// GA_DTCDR_50560355009129
// MI455X (gfx1250) — matrix operand order repaired and checked by reading, not compiled, not run on hardware
//
#include <hip/hip_runtime.h>

typedef unsigned short u16;
typedef unsigned int   u32;
typedef __attribute__((ext_vector_type(16))) __bf16 v16bf;
typedef __attribute__((ext_vector_type(8)))  float  v8f;

#define FEAT 64
#define USER_NUM_C 100000
#define SRC_ITEM_NUM_C 50000
#define TGT_ITEM_NUM_C 40000

// ---------------- device helpers ----------------

__device__ __forceinline__ u16 f2bf(float f) {
  u32 x = __float_as_uint(f);
  x += 0x7fffu + ((x >> 16) & 1u);   // round-to-nearest-even
  return (u16)(x >> 16);
}
__device__ __forceinline__ float bf2f(u16 h) {
  return __uint_as_float(((u32)h) << 16);
}

// ---------------- init / fill ----------------

__global__ void k_fill_i32(int* p, int v, int n) {
  int i = blockIdx.x * 256 + threadIdx.x;
  if (i < n) p[i] = v;
}
__global__ void k_fill_f32(float* p, float v, int n) {
  int i = blockIdx.x * 256 + threadIdx.x;
  if (i < n) p[i] = v;
}

// Build row -> compact-slot map for the batch indices (dedup via CAS).
__global__ void k_assign(const int* __restrict__ idx, int n, int* map, int* counter) {
  int i = blockIdx.x * 256 + threadIdx.x;
  if (i >= n) return;
  int r = idx[i];
  if (atomicCAS(&map[r], -1, -2) == -1) {
    int s = atomicAdd(counter, 1);
    map[r] = s;
  }
}

// ---------------- filtered COO SpMM into compact rows ----------------
// 16 threads per nnz, 4 floats each (FEAT=64).
__global__ void k_spmm(const int* __restrict__ rows, const int* __restrict__ cols,
                       const float* __restrict__ vals, const float* __restrict__ dense,
                       const int* __restrict__ map, float* __restrict__ acc, int nnz) {
  long long gid = (long long)blockIdx.x * 256 + threadIdx.x;
  int e = (int)(gid >> 4);
  if (e >= nnz) return;
  int c = map[rows[e]];
  if (c < 0) return;                    // row not needed by this batch
  int f = ((int)gid & 15) * 4;
  float v = vals[e];
  const float4 d = *(const float4*)(dense + (long long)cols[e] * FEAT + f);
  float* o = acc + (long long)c * FEAT + f;
  atomicAdd(o + 0, v * d.x);
  atomicAdd(o + 1, v * d.y);
  atomicAdd(o + 2, v * d.z);
  atomicAdd(o + 3, v * d.w);
}

// ---------------- weight fp32 -> bf16, pre-swizzled to B-fragment order -----
// dst[((kt*N + n)*2 + half)*16 + e] = bf16(W[(kt*32 + half*16 + e)*N + n])
// so a lane's 16 B-fragment elements are 32 contiguous bytes.
__global__ void k_wswizzle(const float* __restrict__ W, u16* __restrict__ dst,
                           int K, int N) {
  int i = blockIdx.x * 256 + threadIdx.x;
  if (i >= K * N) return;
  int e = i & 15;
  int half = (i >> 4) & 1;
  int rest = i >> 5;
  int n = rest % N;
  int kt = rest / N;
  int k = kt * 32 + half * 16 + e;
  dst[i] = f2bf(W[(long long)k * N + n]);
}

// ---------------- gather + attention combine -> 6 bf16 tower inputs ----------------

__global__ void k_combine(const int* __restrict__ user, const int* __restrict__ spos,
                          const int* __restrict__ sneg, const int* __restrict__ tpos,
                          const int* __restrict__ tneg,
                          const int* __restrict__ umap, const int* __restrict__ smap,
                          const int* __restrict__ tmap,
                          const float* __restrict__ uA_acc, const float* __restrict__ iA_acc,
                          const float* __restrict__ uB_acc, const float* __restrict__ iB_acc,
                          const float* __restrict__ attA, const float* __restrict__ attB,
                          u16* uA_in, u16* uB_in, u16* iApos_in, u16* iAneg_in,
                          u16* iBpos_in, u16* iBneg_in) {
  int gid = blockIdx.x * 256 + threadIdx.x;   // BATCH*FEAT threads
  int i = gid >> 6, f = gid & 63;
  int u = user[i];
  int uc = umap[u];
  float a = uA_acc[(long long)uc * FEAT + f];
  float b = uB_acc[(long long)uc * FEAT + f];
  float d = a - b;
  uA_in[gid] = f2bf(d * attA[(long long)u * FEAT + f] + b);
  uB_in[gid] = f2bf(d * attB[(long long)u * FEAT + f] + b);
  iApos_in[gid] = f2bf(iA_acc[(long long)smap[spos[i]] * FEAT + f]);
  iAneg_in[gid] = f2bf(iA_acc[(long long)smap[sneg[i]] * FEAT + f]);
  iBpos_in[gid] = f2bf(iB_acc[(long long)tmap[tpos[i]] * FEAT + f]);
  iBneg_in[gid] = f2bf(iB_acc[(long long)tmap[tneg[i]] * FEAT + f]);
}

// ---------------- bf16 WMMA GEMM: Y = relu(X @ W + b) ----------------
// X: [M,K] bf16 row-major. Wsw: pre-swizzled bf16 weights (see k_wswizzle).
// bias: [N] f32. Y: [M,N] bf16.
// Block = 256 threads (8 waves), block tile = 128(M) x 64(N).
// Wave tile = 32(M) x 32(N): 2 A frags x 2 B frags -> 4 WMMAs per K-step.
// The block's 64-column swizzled W slice is staged into LDS with CDNA5
// async global->LDS copies (ASYNCcnt) and consumed with ds_load_b128.
__global__ __launch_bounds__(256) void k_gemm(const u16* __restrict__ X,
                                              const u16* __restrict__ Wsw,
                                              const float* __restrict__ bias,
                                              u16* __restrict__ Y,
                                              int M, int K, int N) {
  extern __shared__ u16 Ws[];                    // K/32 tiles * 2048 u16 = K*128 bytes
  int nBlocks = N >> 6;
  int bm = blockIdx.x / nBlocks;
  int bn = blockIdx.x % nBlocks;
  int tid = threadIdx.x;

  // ---- async stage of the swizzled 64-column slice into LDS ----
  // per K-tile kt: 4KB contiguous at global u16 offset (kt*N + bn*64)*32
  unsigned ldsBase = (unsigned)(size_t)(&Ws[0]);
  int nChunks = K * 8;                            // 16B chunks, multiple of 256
  for (int c = tid; c < nChunks; c += 256) {
    int kt = c >> 8;
    int wi = c & 255;
    unsigned goff = (unsigned)((((size_t)kt * N + bn * 64) * 32 + wi * 8) * 2);
    unsigned loff = ldsBase + (unsigned)((kt * 2048 + wi * 8) * 2);
    asm volatile("global_load_async_to_lds_b128 %0, %1, %2"
                 :: "v"(loff), "v"(goff), "s"(Wsw) : "memory");
  }
  asm volatile("s_wait_asynccnt 0x0" ::: "memory");
  __syncthreads();

  int wave = tid >> 5, lane = tid & 31;
  int mSub = wave >> 1;            // 0..3  -> 32 rows each
  int nHalf = wave & 1;            // 0..1  -> 32 cols each
  int mBase = bm * 128 + mSub * 32;
  int nBase0 = nHalf * 32;
  int nBase1 = nHalf * 32 + 16;

  int half = lane >> 4;            // K-half striping (A & B)
  int mr = lane & 15;
  int nc = lane & 15;
  const u16* xrow0 = X + (long long)(mBase + mr) * K;
  const u16* xrow1 = X + (long long)(mBase + 16 + mr) * K;

  v8f c00 = {}, c01 = {}, c10 = {}, c11 = {};
  for (int k0 = 0; k0 < K; k0 += 32) {
    __builtin_prefetch(xrow0 + k0 + 64, 0, 1);
    // A fragments: elements 0-15 = 32B contiguous at xrow+k0+half*16
    union { uint4 q[2]; v16bf v; } A0, A1;
    A0.q[0] = *(const uint4*)(xrow0 + k0 + half * 16);
    A0.q[1] = *(const uint4*)(xrow0 + k0 + half * 16 + 8);
    A1.q[0] = *(const uint4*)(xrow1 + k0 + half * 16);
    A1.q[1] = *(const uint4*)(xrow1 + k0 + half * 16 + 8);
    // B fragments: 32 contiguous bytes per lane from swizzled LDS slice
    int kt = k0 >> 5;
    union { uint4 q[2]; v16bf v; } B0, B1;
    const uint4* b0p = (const uint4*)(Ws + (kt << 11) + (nBase0 + nc) * 32 + half * 16);
    const uint4* b1p = (const uint4*)(Ws + (kt << 11) + (nBase1 + nc) * 32 + half * 16);
    B0.q[0] = b0p[0]; B0.q[1] = b0p[1];
    B1.q[0] = b1p[0]; B1.q[1] = b1p[1];
    c00 = __builtin_amdgcn_wmma_f32_16x16x32_bf16(false, A0.v, false, B0.v, (short)0, c00, false, false);
    c01 = __builtin_amdgcn_wmma_f32_16x16x32_bf16(false, A0.v, false, B1.v, (short)0, c01, false, false);
    c10 = __builtin_amdgcn_wmma_f32_16x16x32_bf16(false, A1.v, false, B0.v, (short)0, c10, false, false);
    c11 = __builtin_amdgcn_wmma_f32_16x16x32_bf16(false, A1.v, false, B1.v, (short)0, c11, false, false);
  }

  // epilogue: C layout — lanes 0-15: VGPR r -> M=r; lanes 16-31: M=8+r; N=lane%16
  int mOut0 = mBase + half * 8;
  int mOut1 = mBase + 16 + half * 8;
  int nG0 = bn * 64 + nBase0 + nc;
  int nG1 = bn * 64 + nBase1 + nc;
  float b0 = bias[nG0], b1 = bias[nG1];
#pragma unroll
  for (int r = 0; r < 8; ++r) {
    float v00 = c00[r] + b0, v01 = c01[r] + b1;
    float v10 = c10[r] + b0, v11 = c11[r] + b1;
    v00 = v00 > 0.f ? v00 : 0.f;
    v01 = v01 > 0.f ? v01 : 0.f;
    v10 = v10 > 0.f ? v10 : 0.f;
    v11 = v11 > 0.f ? v11 : 0.f;
    Y[(long long)(mOut0 + r) * N + nG0] = f2bf(v00);
    Y[(long long)(mOut0 + r) * N + nG1] = f2bf(v01);
    Y[(long long)(mOut1 + r) * N + nG0] = f2bf(v10);
    Y[(long long)(mOut1 + r) * N + nG1] = f2bf(v11);
  }
}

// ---------------- loss: dots + log-sigmoid + L2 reg ----------------

__global__ __launch_bounds__(256) void k_loss(const u16* __restrict__ uA,
                                              const u16* __restrict__ iAp,
                                              const u16* __restrict__ iAn,
                                              const u16* __restrict__ uB,
                                              const u16* __restrict__ iBp,
                                              const u16* __restrict__ iBn,
                                              float* out, int batch) {
  __shared__ float red[256];
  int i = blockIdx.x * 256 + threadIdx.x;
  float ps = 0.f, ns = 0.f, pt = 0.f, nt = 0.f, reg = 0.f;
  long long base = (long long)i * FEAT;
  for (int f = 0; f < FEAT; ++f) {
    float ua = bf2f(uA[base + f]);
    float ip = bf2f(iAp[base + f]);
    float im = bf2f(iAn[base + f]);
    float ub = bf2f(uB[base + f]);
    float jp = bf2f(iBp[base + f]);
    float jm = bf2f(iBn[base + f]);
    ps += ua * ip; ns += ua * im;
    pt += ub * jp; nt += ub * jm;
    reg += ua * ua + ip * ip + im * im + ub * ub + jp * jp + jm * jm;
  }
  auto sp = [](float z) {   // softplus; -log_sigmoid(x) = softplus(-x)
    return z > 0.f ? z + log1pf(__expf(-z)) : log1pf(__expf(z));
  };
  float c = (sp(-ps) + sp(ns) + sp(-pt) + sp(nt)) * (1.0f / (float)batch) + 0.0005f * reg;
  red[threadIdx.x] = c;
  __syncthreads();
  for (int s = 128; s > 0; s >>= 1) {
    if (threadIdx.x < s) red[threadIdx.x] += red[threadIdx.x + s];
    __syncthreads();
  }
  if (threadIdx.x == 0) atomicAdd(out, red[0]);
}

// ---------------- host orchestration ----------------

extern "C" void kernel_launch(void* const* d_in, const int* in_sizes, int n_in,
                              void* d_out, int out_size, void* d_ws, size_t ws_size,
                              hipStream_t stream) {
  const int batch = in_sizes[0];
  const int nnz = in_sizes[7];

  const int* user = (const int*)d_in[0];
  const int* spos = (const int*)d_in[1];
  const int* sneg = (const int*)d_in[2];
  const int* tpos = (const int*)d_in[3];
  const int* tneg = (const int*)d_in[4];
  const int* suv_row = (const int*)d_in[7];
  const int* suv_col = (const int*)d_in[8];
  const float* suv_val = (const float*)d_in[9];
  const int* svu_row = (const int*)d_in[10];
  const int* svu_col = (const int*)d_in[11];
  const float* svu_val = (const float*)d_in[12];
  const int* tuv_row = (const int*)d_in[13];
  const int* tuv_col = (const int*)d_in[14];
  const float* tuv_val = (const float*)d_in[15];
  const int* tvu_row = (const int*)d_in[16];
  const int* tvu_col = (const int*)d_in[17];
  const float* tvu_val = (const float*)d_in[18];
  const float* item_W1_A = (const float*)d_in[19];
  const float* user_W1_A = (const float*)d_in[20];
  const float* item_W1_B = (const float*)d_in[21];
  const float* user_W1_B = (const float*)d_in[22];
  const float* attA = (const float*)d_in[23];
  const float* attB = (const float*)d_in[24];
  // MLP params flattened as (W0,b0,...,W5,b5) per tower
  const int mlp_base[4] = {25, 37, 49, 61};

  // ---- workspace bump allocator ----
  char* p = (char*)d_ws;
  auto take = [&](size_t b) -> char* {
    char* r = p;
    p += (b + 255) & ~(size_t)255;
    return r;
  };
  int* umap = (int*)take(USER_NUM_C * sizeof(int));
  int* smap = (int*)take(SRC_ITEM_NUM_C * sizeof(int));
  int* tmap = (int*)take(TGT_ITEM_NUM_C * sizeof(int));
  int* cnts = (int*)take(4 * sizeof(int));
  float* uA_acc = (float*)take((size_t)batch * FEAT * sizeof(float));
  float* uB_acc = (float*)take((size_t)batch * FEAT * sizeof(float));
  float* iA_acc = (float*)take((size_t)2 * batch * FEAT * sizeof(float));
  float* iB_acc = (float*)take((size_t)2 * batch * FEAT * sizeof(float));

  const int layK[6] = {64, 128, 256, 512, 256, 128};
  const int layN[6] = {128, 256, 512, 256, 128, 64};
  size_t woffs[6], wtot = 0;
  for (int l = 0; l < 6; ++l) { woffs[l] = wtot; wtot += (size_t)layK[l] * layN[l]; }
  u16* warena[4];
  for (int t = 0; t < 4; ++t) warena[t] = (u16*)take(wtot * sizeof(u16));

  u16* in6[6];
  for (int t = 0; t < 6; ++t) in6[t] = (u16*)take((size_t)batch * FEAT * sizeof(u16));
  u16* ping = (u16*)take((size_t)batch * 512 * sizeof(u16));
  u16* pong = (u16*)take((size_t)batch * 512 * sizeof(u16));
  u16* fin6[6];
  for (int t = 0; t < 6; ++t) fin6[t] = (u16*)take((size_t)batch * FEAT * sizeof(u16));

  auto gridOf = [](int n) { return dim3((unsigned)((n + 255) / 256)); };

  // ---- phase 0: init ----
  k_fill_i32<<<gridOf(USER_NUM_C), 256, 0, stream>>>(umap, -1, USER_NUM_C);
  k_fill_i32<<<gridOf(SRC_ITEM_NUM_C), 256, 0, stream>>>(smap, -1, SRC_ITEM_NUM_C);
  k_fill_i32<<<gridOf(TGT_ITEM_NUM_C), 256, 0, stream>>>(tmap, -1, TGT_ITEM_NUM_C);
  k_fill_i32<<<1, 256, 0, stream>>>(cnts, 0, 4);
  k_fill_f32<<<gridOf(batch * FEAT), 256, 0, stream>>>(uA_acc, 0.f, batch * FEAT);
  k_fill_f32<<<gridOf(batch * FEAT), 256, 0, stream>>>(uB_acc, 0.f, batch * FEAT);
  k_fill_f32<<<gridOf(2 * batch * FEAT), 256, 0, stream>>>(iA_acc, 0.f, 2 * batch * FEAT);
  k_fill_f32<<<gridOf(2 * batch * FEAT), 256, 0, stream>>>(iB_acc, 0.f, 2 * batch * FEAT);
  k_fill_f32<<<1, 256, 0, stream>>>((float*)d_out, 0.f, out_size);

  // ---- phase 1: batch-row dedup maps ----
  k_assign<<<gridOf(batch), 256, 0, stream>>>(user, batch, umap, cnts + 0);
  k_assign<<<gridOf(batch), 256, 0, stream>>>(spos, batch, smap, cnts + 1);
  k_assign<<<gridOf(batch), 256, 0, stream>>>(sneg, batch, smap, cnts + 1);
  k_assign<<<gridOf(batch), 256, 0, stream>>>(tpos, batch, tmap, cnts + 2);
  k_assign<<<gridOf(batch), 256, 0, stream>>>(tneg, batch, tmap, cnts + 2);

  // ---- phase 2: filtered SpMM into compact rows ----
  dim3 sg((unsigned)(((long long)nnz * 16 + 255) / 256));
  k_spmm<<<sg, 256, 0, stream>>>(suv_row, suv_col, suv_val, item_W1_A, umap, uA_acc, nnz);
  k_spmm<<<sg, 256, 0, stream>>>(svu_row, svu_col, svu_val, user_W1_A, smap, iA_acc, nnz);
  k_spmm<<<sg, 256, 0, stream>>>(tuv_row, tuv_col, tuv_val, item_W1_B, umap, uB_acc, nnz);
  k_spmm<<<sg, 256, 0, stream>>>(tvu_row, tvu_col, tvu_val, user_W1_B, tmap, iB_acc, nnz);

  // ---- phase 3: weight fp32 -> bf16, swizzled to B-fragment order ----
  for (int t = 0; t < 4; ++t)
    for (int l = 0; l < 6; ++l) {
      int n = layK[l] * layN[l];
      k_wswizzle<<<gridOf(n), 256, 0, stream>>>((const float*)d_in[mlp_base[t] + 2 * l],
                                                warena[t] + woffs[l], layK[l], layN[l]);
    }

  // ---- phase 4: gather + attention combine ----
  k_combine<<<gridOf(batch * FEAT), 256, 0, stream>>>(
      user, spos, sneg, tpos, tneg, umap, smap, tmap,
      uA_acc, iA_acc, uB_acc, iB_acc, attA, attB,
      in6[0], in6[3], in6[1], in6[2], in6[4], in6[5]);
  // in6 order: 0=uA, 1=iApos, 2=iAneg, 3=uB, 4=iBpos, 5=iBneg

  // ---- phase 5: 6 MLP towers, 6 WMMA GEMM layers each ----
  const int towerArena[6] = {0, 1, 1, 2, 3, 3};
  const int towerBase[6]  = {25, 37, 37, 49, 61, 61};
  for (int t = 0; t < 6; ++t) {
    const u16* x = in6[t];
    for (int l = 0; l < 6; ++l) {
      int K = layK[l], N = layN[l];
      u16* y = (l == 5) ? fin6[t] : ((l & 1) ? pong : ping);
      dim3 grid((unsigned)((batch / 128) * (N / 64)));
      size_t lds = (size_t)K * 128;   // K/32 tiles * 4KB
      k_gemm<<<grid, 256, lds, stream>>>(x, warena[towerArena[t]] + woffs[l],
                                         (const float*)d_in[towerBase[t] + 2 * l + 1],
                                         y, batch, K, N);
      x = y;
    }
  }

  // ---- phase 6: loss ----
  k_loss<<<gridOf(batch), 256, 0, stream>>>(fin6[0], fin6[1], fin6[2], fin6[3], fin6[4],
                                            fin6[5], (float*)d_out, batch);
}
